// MultiHeadAttention_29111288333143
// MI455X (gfx1250) — matrix operand order repaired and checked by reading, not compiled, not run on hardware
//
#include <hip/hip_runtime.h>

typedef __attribute__((ext_vector_type(16))) __bf16 v16bf;
typedef __attribute__((ext_vector_type(8)))  __bf16 v8bf;
typedef __attribute__((ext_vector_type(8)))  float  v8f;

#define B_  2
#define H_  12
#define N_  2048
#define D_  768
#define HD_ 64
#define BH_ (B_ * H_)

// ---------- helpers ----------

// native hardware conversion (v_cvt_pk_bf16_f32 on gfx1250)
static __device__ __forceinline__ __bf16 f2bf(float f) { return (__bf16)f; }

static __device__ __forceinline__ v8f wmma_bf16(v16bf a, v16bf b, v8f c) {
  return __builtin_amdgcn_wmma_f32_16x16x32_bf16(false, a, false, b, (short)0, c,
                                                 false, false);
}

// two contiguous 8-bf16 runs -> v16bf (lowers to 2x b128 loads)
static __device__ __forceinline__ v16bf ld8x2(const __bf16* p0, const __bf16* p1) {
  v8bf lo = *(const v8bf*)p0;
  v8bf hi = *(const v8bf*)p1;
  v16bf r;
#pragma unroll
  for (int i = 0; i < 8; ++i) { r[i] = lo[i]; r[i + 8] = hi[i]; }
  return r;
}
static __device__ __forceinline__ v16bf ld16(const __bf16* p) { return ld8x2(p, p + 8); }

static __device__ __forceinline__ float redmax16(float v) {
  v = fmaxf(v, __shfl_xor(v, 1, 32));
  v = fmaxf(v, __shfl_xor(v, 2, 32));
  v = fmaxf(v, __shfl_xor(v, 4, 32));
  v = fmaxf(v, __shfl_xor(v, 8, 32));
  return v;
}
static __device__ __forceinline__ float redsum16(float v) {
  v += __shfl_xor(v, 1, 32);
  v += __shfl_xor(v, 2, 32);
  v += __shfl_xor(v, 4, 32);
  v += __shfl_xor(v, 8, 32);
  return v;
}

// ---------- kernel 0: streaming f32 -> bf16 conversion (one pass) ----------

__global__ __launch_bounds__(256)
void cvt_bf16_kernel(const float* __restrict__ in, __bf16* __restrict__ out, int n8) {
  const int i = blockIdx.x * blockDim.x + threadIdx.x;
  if (i >= n8) return;
  const float4 f0 = *(const float4*)(in + (size_t)i * 8);
  const float4 f1 = *(const float4*)(in + (size_t)i * 8 + 4);
  v8bf r;
  r[0] = f2bf(f0.x); r[1] = f2bf(f0.y); r[2] = f2bf(f0.z); r[3] = f2bf(f0.w);
  r[4] = f2bf(f1.x); r[5] = f2bf(f1.y); r[6] = f2bf(f1.z); r[7] = f2bf(f1.w);
  *(v8bf*)(out + (size_t)i * 8) = r;
}

// ---------- kernel 1: fused QKV projection, fully unrolled ----------
// wave -> one 16x64 strip of one projection.  The 24-step K loop is fully
// unrolled: no back-edge, so the scheduler pipelines loads ahead of WMMAs
// with a sliding register window and zero rotation copies.

__global__ __launch_bounds__(128)
void qkv_proj_kernel(const __bf16* __restrict__ xb,
                     const __bf16* __restrict__ Wqb,
                     const __bf16* __restrict__ Wkb,
                     const __bf16* __restrict__ Wvb,
                     __bf16* __restrict__ Qo,
                     __bf16* __restrict__ Ko,
                     __bf16* __restrict__ Vt) {
  const int lane = threadIdx.x & 31;
  const int lh = lane >> 4;          // 0/1 half-wave
  const int ll = lane & 15;
  const int w = blockIdx.x * 4 + (threadIdx.x >> 5);

  const int MT = (B_ * N_) / 16;     // 256 row tiles
  const int NG = D_ / 64;            // 12 col strips
  const int p = w / (MT * NG);       // 0=Q 1=K 2=V
  int rem = w - p * MT * NG;
  const int mt = rem / NG;
  const int ng = rem % NG;
  const __bf16* W = (p == 0) ? Wqb : (p == 1) ? Wkb : Wvb;

  v8f acc[4] = {};
  const __bf16* arow = xb + (size_t)(mt * 16 + ll) * D_ + lh * 16;
  const __bf16* wbase = W + (size_t)(ng * 64 + ll) * D_ + lh * 16;

#pragma unroll
  for (int kk = 0; kk < D_; kk += 32) {      // 24 straight-line k-steps
    v16bf a = ld16(arow + kk);
#pragma unroll
    for (int t = 0; t < 4; ++t) {
      v16bf b = ld16(wbase + (size_t)t * 16 * D_ + kk);
      acc[t] = wmma_bf16(a, b, acc[t]);
    }
  }

#pragma unroll
  for (int t = 0; t < 4; ++t) {
#pragma unroll
    for (int r = 0; r < 8; ++r) {
      const int tok = mt * 16 + r + 8 * lh;
      const int bb = tok >> 11, nn = tok & (N_ - 1);
      const int ncol = ng * 64 + t * 16 + ll;
      const int hh = ncol >> 6, hd = ncol & 63;
      const __bf16 v = f2bf(acc[t][r]);
      const size_t bh = (size_t)bb * H_ + hh;
      if (p == 0)      Qo[(bh * N_ + nn) * HD_ + hd] = v;
      else if (p == 1) Ko[(bh * N_ + nn) * HD_ + hd] = v;
      else             Vt[(bh * HD_ + hd) * N_ + nn] = v;   // transposed
    }
  }
}

// ---------- kernel 2: causal flash attention, one wave per 16-query tile ----------

__global__ __launch_bounds__(32)
void attn_kernel(const __bf16* __restrict__ Q,
                 const __bf16* __restrict__ K,
                 const __bf16* __restrict__ Vt,
                 float* __restrict__ out) {
  __shared__ __attribute__((aligned(16))) __bf16 ldsP[16 * 32];

  const int lane = threadIdx.x & 31;
  const int lh = lane >> 4;
  const int ll = lane & 15;
  const int tile = blockIdx.x & 127;        // 128 q-tiles of 16
  const int bh = blockIdx.x >> 7;           // 0..23
  const int q0 = tile * 16;

  const __bf16* Qb = Q + (size_t)bh * N_ * HD_;
  const __bf16* Kb = K + (size_t)bh * N_ * HD_;
  const __bf16* Vb = Vt + (size_t)bh * HD_ * N_;

  // Q A-fragments (d-chunks 0..31, 32..63), loaded once
  v16bf qa[2];
#pragma unroll
  for (int c = 0; c < 2; ++c) {
    const __bf16* qp = Qb + (size_t)(q0 + ll) * HD_ + c * 32 + lh * 16;
    qa[c] = ld16(qp);
  }

  v8f o[4] = {};
  float m[8], l[8];
#pragma unroll
  for (int r = 0; r < 8; ++r) { m[r] = -1e30f; l[r] = 0.0f; }

  const int steps = (q0 + 16 + 31) >> 5;    // causal: only kv blocks that matter
  for (int j = 0; j < steps; ++j) {
    const int kvb = j * 32;

    // ---- issue ALL loads for this kv block up front: K fragments (needed
    // soon) and V fragments (needed only after the softmax chain, so their
    // latency hides under exp/shfl work) ----
    v16bf kb[4];   // [c*2 + t]
#pragma unroll
    for (int c = 0; c < 2; ++c) {
      kb[c * 2 + 0] = ld16(Kb + (size_t)(kvb + ll) * HD_ + c * 32 + lh * 16);
      kb[c * 2 + 1] = ld16(Kb + (size_t)(kvb + 16 + ll) * HD_ + c * 32 + lh * 16);
    }
    v16bf vb[4];
#pragma unroll
    for (int t = 0; t < 4; ++t)
      vb[t] = ld16(Vb + (size_t)(t * 16 + ll) * N_ + kvb + lh * 16);

    // ---- scores: two 16x16 tiles, K-dim 64 = 2 chunks ----
    v8f s0 = {}, s1 = {};
#pragma unroll
    for (int c = 0; c < 2; ++c) {
      s0 = wmma_bf16(qa[c], kb[c * 2 + 0], s0);
      s1 = wmma_bf16(qa[c], kb[c * 2 + 1], s1);
    }

    // ---- causal mask + online softmax ----
    const int col0 = kvb + ll, col1 = kvb + 16 + ll;
    float al[8], p0[8], p1[8];
#pragma unroll
    for (int r = 0; r < 8; ++r) {
      const int qrow = q0 + r + 8 * lh;
      const float v0 = (col0 <= qrow) ? s0[r] : -1e30f;
      const float v1 = (col1 <= qrow) ? s1[r] : -1e30f;
      const float rowmax = redmax16(fmaxf(v0, v1));
      const float mn = fmaxf(m[r], rowmax);
      al[r] = __expf(m[r] - mn);
      p0[r] = __expf(v0 - mn);
      p1[r] = __expf(v1 - mn);
      const float rs = redsum16(p0[r] + p1[r]);
      l[r] = l[r] * al[r] + rs;
      m[r] = mn;
    }
#pragma unroll
    for (int t = 0; t < 4; ++t)
#pragma unroll
      for (int r = 0; r < 8; ++r) o[t][r] *= al[r];

    // ---- P: C-layout -> LDS row-major 16x32 -> A-fragment ----
#pragma unroll
    for (int r = 0; r < 8; ++r) {
      const int row = r + 8 * lh;
      ldsP[row * 32 + ll]      = f2bf(p0[r]);
      ldsP[row * 32 + 16 + ll] = f2bf(p1[r]);
    }
    __syncthreads();                       // single-wave WG: HW NOP, compiler fence
    const __bf16* lp = ldsP + ll * 32 + lh * 16;
    v16bf pa = ld16(lp);
    __syncthreads();

    // ---- O += P * V (V fragments already in registers) ----
#pragma unroll
    for (int t = 0; t < 4; ++t) o[t] = wmma_bf16(pa, vb[t], o[t]);
  }

  // ---- normalize and store [b, n, h*64+hd] f32 ----
  const int bb = bh / H_, hh = bh % H_;
  float inv[8];
#pragma unroll
  for (int r = 0; r < 8; ++r) inv[r] = 1.0f / l[r];
#pragma unroll
  for (int t = 0; t < 4; ++t)
#pragma unroll
    for (int r = 0; r < 8; ++r) {
      const int qrow = q0 + r + 8 * lh;
      out[((size_t)(bb * N_ + qrow)) * D_ + hh * 64 + t * 16 + ll] = o[t][r] * inv[r];
    }
}

// ---------- launch ----------

extern "C" void kernel_launch(void* const* d_in, const int* in_sizes, int n_in,
                              void* d_out, int out_size, void* d_ws, size_t ws_size,
                              hipStream_t stream) {
  const float* x  = (const float*)d_in[0];
  const float* Wq = (const float*)d_in[1];
  const float* Wk = (const float*)d_in[2];
  const float* Wv = (const float*)d_in[3];

  const size_t elems = (size_t)BH_ * N_ * HD_;   // 3,145,728 (= B*N*D)
  const size_t welems = (size_t)D_ * D_;         // 589,824

  __bf16* q   = (__bf16*)d_ws;
  __bf16* k   = q + elems;
  __bf16* vt  = k + elems;
  __bf16* xb  = vt + elems;
  __bf16* wqb = xb + elems;
  __bf16* wkb = wqb + welems;
  __bf16* wvb = wkb + welems;

  // one-pass f32 -> bf16 pre-conversion (streaming, bandwidth bound)
  {
    const int n8x = (int)(elems / 8);            // 393,216
    const int n8w = (int)(welems / 8);           // 73,728
    cvt_bf16_kernel<<<(n8x + 255) / 256, 256, 0, stream>>>(x,  xb,  n8x);
    cvt_bf16_kernel<<<(n8w + 255) / 256, 256, 0, stream>>>(Wq, wqb, n8w);
    cvt_bf16_kernel<<<(n8w + 255) / 256, 256, 0, stream>>>(Wk, wkb, n8w);
    cvt_bf16_kernel<<<(n8w + 255) / 256, 256, 0, stream>>>(Wv, wvb, n8w);
  }

  // 3 * 256 * 12 = 9216 waves, 4 waves / block
  qkv_proj_kernel<<<9216 / 4, 128, 0, stream>>>(xb, wqb, wkb, wvb, q, k, vt);
  // 24 bh * 128 q-tiles, 1 wave / block
  attn_kernel<<<BH_ * 128, 32, 0, stream>>>(q, k, vt, (float*)d_out);
}
